// VQVAE_90640989815346
// MI455X (gfx1250) — matrix operand order repaired and checked by reading, not compiled, not run on hardware
//
#include <hip/hip_runtime.h>
#include <hip/hip_bf16.h>

// ---------------------------------------------------------------------------
// VQ-VAE forward for MI455X (gfx1250), wave32 + v_wmma_f32_16x16x32_f16.
// Compute-bound (~340 GFLOP vs ~50MB traffic) -> f16-in/f32-acc WMMA path.
// VQ faithful-to-source: argmin over size-1 axis -> index 0 always, so
// z_q == codebook[0] broadcast, perplexity == 1.0f (f32 arithmetic).
// ---------------------------------------------------------------------------

typedef _Float16 h16;
typedef __attribute__((ext_vector_type(16))) _Float16 v16h;
typedef __attribute__((ext_vector_type(8)))  _Float16 v8h;
typedef __attribute__((ext_vector_type(8)))  float    v8f;

#define DEVINL __device__ __forceinline__

#define NB    32
#define NT    128
#define NMOD  53
#define BT    4096           // NB*NT
#define ROWS  217088         // BT*NMOD
#define HDIM  256
#define LDIM  64
#define RECON_ELEMS 651264   // ROWS*3

// --- WMMA fragment helpers (contiguous k order, same for A and B) ----------------------------
// A 16x32 f16: lane l -> M=l%16; element e -> K = 16*g + e
//   => one contiguous 16-half run [16g, 16g+16).
// B 32x16 f16: lane l -> N=l%16; element e -> K = 16*g + e
//   => one contiguous 16-half run [16g, 16g+16). We store B transposed [N][K].
DEVINL v16h frag_a(const h16* row, int g, int kc) {
  v8h lo = *(const v8h*)(row + 16*g + 32*kc);
  v8h hi = *(const v8h*)(row + 16*g + 32*kc + 8);
  v16h a;
#pragma unroll
  for (int i = 0; i < 8; ++i) { a[i] = lo[i]; a[i+8] = hi[i]; }
  return a;
}
DEVINL v16h frag_b(const h16* row, int g, int kc) {
  v8h lo = *(const v8h*)(row + 16*g + 32*kc);
  v8h hi = *(const v8h*)(row + 16*g + 32*kc + 8);
  v16h b;
#pragma unroll
  for (int i = 0; i < 8; ++i) { b[i] = lo[i]; b[i+8] = hi[i]; }
  return b;
}
DEVINL v8f wmma16(v16h a, v16h b, v8f c) {
  return __builtin_amdgcn_wmma_f32_16x16x32_f16(false, a, false, b,
                                                (short)0, c, false, false);
}

// --- generic WMMA GEMM: C[M,N] = A[M,K](f16) @ BT[N,K](f16) (+bias, relu) --
// one wave computes a 16x64 C strip; M%16==0, N%64==0, K%32==0.
template<bool RELU, bool BIAS, bool OUTF16>
__global__ __launch_bounds__(256) void gemm_wmma(
    const h16* __restrict__ A, const h16* __restrict__ Bt,
    const float* __restrict__ bias, void* __restrict__ C,
    int M, int N, int K) {
  const int lane = threadIdx.x & 31;
  const int wid  = blockIdx.x * 8 + (threadIdx.x >> 5);
  const int mtiles = M >> 4;
  if (wid >= mtiles * (N >> 6)) return;
  const int m0 = (wid % mtiles) << 4;
  const int n0 = (wid / mtiles) << 6;
  const int g = lane >> 4, ln = lane & 15;
  v8f acc[4] = {};
  const h16* Arow = A + (size_t)(m0 + ln) * K;
  const int nkc = K >> 5;
  for (int kc = 0; kc < nkc; ++kc) {
    v16h a = frag_a(Arow, g, kc);
#pragma unroll
    for (int nt = 0; nt < 4; ++nt) {
      v16h b = frag_b(Bt + (size_t)(n0 + nt*16 + ln) * K, g, kc);
      acc[nt] = wmma16(a, b, acc[nt]);
    }
  }
#pragma unroll
  for (int nt = 0; nt < 4; ++nt) {
    const int col = n0 + nt*16 + ln;
    const float bv = BIAS ? bias[col] : 0.0f;
#pragma unroll
    for (int r = 0; r < 8; ++r) {
      const int row = m0 + r + 8*g;
      float v = acc[nt][r] + bv;
      if (RELU) v = fmaxf(v, 0.0f);
      if (OUTF16) ((h16*)C)[(size_t)row * N + col] = (h16)v;
      else        ((float*)C)[(size_t)row * N + col] = v;
    }
  }
}

// --- fused qkv -> attention -> mean -> out-proj, one block per bt ----------
#define LDH   264   // 256+8 pad (halves)
#define LDQ   776   // 768+8 pad
#define LDATT 40    // 32+8 pad
__global__ __launch_bounds__(256) void enc_attn(
    const h16* __restrict__ h2, const h16* __restrict__ qkv_t,
    const float* __restrict__ out_w, const float* __restrict__ out_b,
    float* __restrict__ hbar) {
  __shared__ __align__(16) h16 sH[64 * LDH];     //  33.0 KB
  __shared__ __align__(16) h16 sQ[64 * LDQ];     //  97.0 KB (4 heads/pass)
  __shared__ __align__(16) h16 sAt[8 * 32 * LDATT]; // 20 KB
  __shared__ float sOb[512];                     //   2 KB   total ~152 KB
  const int bt = blockIdx.x;
  const int tid = threadIdx.x;
  const int lane = tid & 31, widx = tid >> 5;
  const int g = lane >> 4, ln = lane & 15;

  for (int i = tid; i < 64 * 256; i += 256) {
    const int r = i >> 8, c = i & 255;
    sH[r*LDH + c] = (r < NMOD) ? h2[((size_t)bt*NMOD + r)*HDIM + c] : (h16)0.0f;
  }
  if (tid < 512) sOb[tid] = 0.0f;
  __syncthreads();

  for (int p = 0; p < 2; ++p) {   // heads 4p..4p+3
    // ---- qkv GEMM slab: local cols = seg*256 + hl*64 + d (seg: q/k/v) ----
#pragma unroll 1
    for (int ntg = 0; ntg < 6; ++ntg) {
      const int nt  = widx*6 + ntg;                 // 0..47 local n-tile
      const int seg = nt >> 4;
      const int gc0 = seg*512 + p*256 + (nt & 15)*16; // global qkv column
      v8f acc[4] = {};
      const h16* Brow = qkv_t + (size_t)(gc0 + ln) * 256;
#pragma unroll
      for (int kc = 0; kc < 8; ++kc) {
        v16h b = frag_b(Brow, g, kc);
#pragma unroll
        for (int mi = 0; mi < 4; ++mi) {
          v16h a = frag_a(sH + (mi*16 + ln)*LDH, g, kc);
          acc[mi] = wmma16(a, b, acc[mi]);
        }
      }
#pragma unroll
      for (int mi = 0; mi < 4; ++mi)
#pragma unroll
        for (int r = 0; r < 8; ++r)
          sQ[(mi*16 + r + 8*g)*LDQ + nt*16 + ln] = (h16)acc[mi][r];
    }
    __syncthreads();

    // ---- attention: 2 waves/head, wave handles rows 32u..32u+31 ----------
    const int hl = widx >> 1, u = widx & 1;
    const int qo = hl*64, ko = 256 + hl*64, vo = 512 + hl*64;
    v8f D[2][4] = {};
#pragma unroll
    for (int kc = 0; kc < 2; ++kc) {
      v16h aq[2];
#pragma unroll
      for (int m = 0; m < 2; ++m)
        aq[m] = frag_a(sQ + ((2*u+m)*16 + ln)*LDQ + qo, g, kc);
#pragma unroll
      for (int nj = 0; nj < 4; ++nj) {
        v16h bk = frag_b(sQ + (nj*16 + ln)*LDQ + ko, g, kc); // k^T: contiguous
#pragma unroll
        for (int m = 0; m < 2; ++m) D[m][nj] = wmma16(aq[m], bk, D[m][nj]);
      }
    }
    // scale + pad-column mask
#pragma unroll
    for (int m = 0; m < 2; ++m)
#pragma unroll
      for (int nj = 0; nj < 4; ++nj) {
        const int col = nj*16 + ln;
#pragma unroll
        for (int r = 0; r < 8; ++r) {
          const float v = D[m][nj][r] * 0.125f;   // DH^-0.5
          D[m][nj][r] = (col < NMOD) ? v : -1e30f;
        }
      }
    // row softmax: each row lives in VGPR r of one 16-lane group
#pragma unroll
    for (int m = 0; m < 2; ++m)
#pragma unroll
      for (int r = 0; r < 8; ++r) {
        float mx = fmaxf(fmaxf(D[m][0][r], D[m][1][r]),
                         fmaxf(D[m][2][r], D[m][3][r]));
#pragma unroll
        for (int off = 1; off < 16; off <<= 1)
          mx = fmaxf(mx, __shfl_xor(mx, off, 16));
        float s = 0.0f;
#pragma unroll
        for (int nj = 0; nj < 4; ++nj) {
          const float e = __expf(D[m][nj][r] - mx);
          D[m][nj][r] = e; s += e;
        }
#pragma unroll
        for (int off = 1; off < 16; off <<= 1) s += __shfl_xor(s, off, 16);
        const float inv = 1.0f / s;
#pragma unroll
        for (int nj = 0; nj < 4; ++nj) D[m][nj][r] *= inv;
      }
    // attn @ v : stage attn chunk per wave, v gathered per element from LDS
    h16* sA = sAt + widx * 32 * LDATT;
    v8f O[2][4] = {};
#pragma unroll 1
    for (int kc = 0; kc < 2; ++kc) {
#pragma unroll
      for (int m = 0; m < 2; ++m)
#pragma unroll
        for (int jc = 0; jc < 2; ++jc)
#pragma unroll
          for (int r = 0; r < 8; ++r)
            sA[(m*16 + r + 8*g)*LDATT + jc*16 + ln] = (h16)D[m][2*kc+jc][r];
      v16h aa[2];
#pragma unroll
      for (int m = 0; m < 2; ++m)
        aa[m] = frag_a(sA + (m*16 + ln)*LDATT, g, 0);
#pragma unroll
      for (int nd = 0; nd < 4; ++nd) {
        v16h bv;
        const h16* vc = sQ + vo + nd*16 + ln;
#pragma unroll
        for (int e = 0; e < 16; ++e)
          bv[e] = vc[(size_t)(16*g + e + 32*kc) * LDQ];
#pragma unroll
        for (int m = 0; m < 2; ++m) O[m][nd] = wmma16(aa[m], bv, O[m][nd]);
      }
    }
    // mean-before-linear: accumulate column sums of o over rows<53
#pragma unroll
    for (int nd = 0; nd < 4; ++nd) {
      float s = 0.0f;
#pragma unroll
      for (int m = 0; m < 2; ++m)
#pragma unroll
        for (int r = 0; r < 8; ++r) {
          const int row = (2*u+m)*16 + r + 8*g;
          if (row < NMOD) s += O[m][nd][r];
        }
      s += __shfl_xor(s, 16, 32);
      if (g == 0) atomicAdd(&sOb[p*256 + hl*64 + nd*16 + ln], s);
    }
    __syncthreads();
  }
  // out-proj on the 512-vector mean (fp32 weights, tiny)
  {
    const int c = tid;
    float s = out_b[c];
    for (int k = 0; k < 512; ++k)
      s += (sOb[k] * (1.0f / 53.0f)) * out_w[(size_t)k*HDIM + c];
    hbar[(size_t)bt*HDIM + c] = s;
  }
}

// --- small helper kernels ---------------------------------------------------
__global__ void init_acc_k(float* acc) { acc[0] = 0.0f; }

__global__ void transpose_h_k(const float* __restrict__ src, h16* __restrict__ dst,
                              int K, int N) {  // src [K,N] -> dst [N,K] fp16
  const int i = blockIdx.x*256 + threadIdx.x;
  if (i >= K*N) return;
  const int n = i / K, k = i - n*K;
  dst[i] = (h16)src[(size_t)k*N + n];
}
__global__ void convw_h_k(const float* __restrict__ w, h16* __restrict__ dst) {
  // w [256][256][3] -> Bt [o][kk*256+i]
  const int idx = blockIdx.x*256 + threadIdx.x;     // 256*768
  const int o = idx / 768, rem = idx - o*768;
  const int kk = rem >> 8, ci = rem & 255;
  dst[idx] = (h16)w[(size_t)o*768 + ci*3 + kk];
}
__global__ void mlp1_k(const float* __restrict__ x, const float* __restrict__ w1,
                       const float* __restrict__ b1, h16* __restrict__ h1) {
  const size_t i = (size_t)blockIdx.x*256 + threadIdx.x;
  const size_t r = i >> 8; const int c = (int)(i & 255);
  const float* xr = x + r*3;
  const float v = b1[c] + xr[0]*w1[c] + xr[1]*w1[256+c] + xr[2]*w1[512+c];
  h1[i] = (h16)fmaxf(v, 0.0f);
}
__global__ void im2col_f32_k(const float* __restrict__ src, h16* __restrict__ dst) {
  const int i = blockIdx.x*256 + threadIdx.x;       // BT*768
  const int bt = i / 768, rem = i - bt*768;
  const int kk = rem >> 8, c = rem & 255;
  const int b = bt >> 7, t = bt & 127, ts = t + kk - 1;
  dst[i] = (ts >= 0 && ts < NT) ? (h16)src[((size_t)(b*NT + ts))*HDIM + c] : (h16)0.0f;
}
__global__ void im2col_f16_k(const h16* __restrict__ src, h16* __restrict__ dst) {
  const int i = blockIdx.x*256 + threadIdx.x;
  const int bt = i / 768, rem = i - bt*768;
  const int kk = rem >> 8, c = rem & 255;
  const int b = bt >> 7, t = bt & 127, ts = t + kk - 1;
  dst[i] = (ts >= 0 && ts < NT) ? src[((size_t)(b*NT + ts))*HDIM + c] : (h16)0.0f;
}
__global__ void im2col_const_k(const float* __restrict__ dconst, h16* __restrict__ dst) {
  const int i = blockIdx.x*256 + threadIdx.x;
  const int bt = i / 768, rem = i - bt*768;
  const int kk = rem >> 8, c = rem & 255;
  const int t = bt & 127, ts = t + kk - 1;
  dst[i] = (ts >= 0 && ts < NT) ? (h16)dconst[c] : (h16)0.0f;
}
__global__ void dconst_k(const float* __restrict__ cb, const float* __restrict__ w,
                         const float* __restrict__ b, float* __restrict__ dconst) {
  const int c = threadIdx.x;                        // 256
  float s = b[c];
  for (int l = 0; l < LDIM; ++l) s += cb[l] * w[(size_t)l*HDIM + c];
  dconst[c] = s;
}
__global__ void seedadd_k(const h16* __restrict__ ddec, const float* __restrict__ seeds,
                          h16* __restrict__ p0) {
  const size_t i = (size_t)blockIdx.x*256 + threadIdx.x;  // ROWS*256
  const size_t r = i >> 8; const int c = (int)(i & 255);
  const int bt = (int)(r / NMOD);
  p0[i] = (h16)((float)ddec[(size_t)bt*HDIM + c] + seeds[i]);
}
__global__ void gen3_k(const h16* __restrict__ p, const float* __restrict__ w3,
                       const float* __restrict__ b3, float* __restrict__ out) {
  const size_t i = (size_t)blockIdx.x*256 + threadIdx.x;  // ROWS*3
  const size_t r = i / 3; const int j = (int)(i - r*3);
  const h16* pr = p + r*HDIM;
  float s = b3[j];
  for (int c = 0; c < HDIM; ++c) s += (float)pr[c] * w3[c*3 + j];
  out[i] = s;
}
__global__ void vqloss_k(const float* __restrict__ z, const float* __restrict__ cb,
                         float* __restrict__ acc) {
  __shared__ float red[256];
  const int i = blockIdx.x*256 + threadIdx.x;       // BT*64
  const float d = z[i] - cb[i & 63];
  red[threadIdx.x] = d * d;
  __syncthreads();
  for (int s = 128; s > 0; s >>= 1) {
    if (threadIdx.x < s) red[threadIdx.x] += red[threadIdx.x + s];
    __syncthreads();
  }
  if (threadIdx.x == 0) atomicAdd(acc, red[0]);
}
__global__ void finalize_k(const float* __restrict__ acc, float* __restrict__ out2) {
  out2[0] = 1.25f * acc[0] / 262144.0f;  // q_loss + 0.25*e_loss, same MSE
  out2[1] = 1.0f;                        // perplexity (f32 exact)
}

// ---------------------------------------------------------------------------
extern "C" void kernel_launch(void* const* d_in, const int* in_sizes, int n_in,
                              void* d_out, int out_size, void* d_ws, size_t ws_size,
                              hipStream_t stream) {
  (void)in_sizes; (void)n_in; (void)out_size; (void)ws_size;
  const float* x      = (const float*)d_in[0];
  const float* seeds  = (const float*)d_in[1];
  const float* mlp_w1 = (const float*)d_in[2];
  const float* mlp_b1 = (const float*)d_in[3];
  const float* mlp_w2 = (const float*)d_in[4];
  const float* mlp_b2 = (const float*)d_in[5];
  const float* qkv_w  = (const float*)d_in[6];
  const float* out_w  = (const float*)d_in[7];
  const float* out_b  = (const float*)d_in[8];
  const float* ec1_w  = (const float*)d_in[9];
  const float* ec1_b  = (const float*)d_in[10];
  const float* ec2_w  = (const float*)d_in[11];
  const float* ec2_b  = (const float*)d_in[12];
  const float* proj_w = (const float*)d_in[13];
  const float* proj_b = (const float*)d_in[14];
  const float* cb     = (const float*)d_in[15];
  const float* di_w   = (const float*)d_in[16];
  const float* di_b   = (const float*)d_in[17];
  const float* dc1_w  = (const float*)d_in[18];
  const float* dc1_b  = (const float*)d_in[19];
  const float* dc2_w  = (const float*)d_in[20];
  const float* dc2_b  = (const float*)d_in[21];
  const float* g1_w   = (const float*)d_in[22];
  const float* g1_b   = (const float*)d_in[23];
  const float* g2_w   = (const float*)d_in[24];
  const float* g2_b   = (const float*)d_in[25];
  const float* g3_w   = (const float*)d_in[26];
  const float* g3_b   = (const float*)d_in[27];
  float* out = (float*)d_out;

  char* ws = (char*)d_ws;
  size_t off = 0;
  auto alloc = [&](size_t b) { size_t r = off; off += (b + 255) & ~(size_t)255; return r; };
  h16*   mlp2_t = (h16*)(ws + alloc(256*256*2));
  h16*   qkv_t  = (h16*)(ws + alloc(1536*256*2));
  h16*   proj_t = (h16*)(ws + alloc(64*256*2));
  h16*   g1_t   = (h16*)(ws + alloc(256*256*2));
  h16*   g2_t   = (h16*)(ws + alloc(256*256*2));
  h16*   ec1_t  = (h16*)(ws + alloc(256*768*2));
  h16*   ec2_t  = (h16*)(ws + alloc(256*768*2));
  h16*   dc1_t  = (h16*)(ws + alloc(256*768*2));
  h16*   dc2_t  = (h16*)(ws + alloc(256*768*2));
  h16*   h1     = (h16*)(ws + alloc((size_t)ROWS*256*2));   // also p0
  h16*   h2     = (h16*)(ws + alloc((size_t)ROWS*256*2));   // also p1
  float* hbar   = (float*)(ws + alloc((size_t)BT*256*4));
  h16*   ic     = (h16*)(ws + alloc((size_t)BT*768*2));
  h16*   c1     = (h16*)(ws + alloc((size_t)BT*256*2));
  h16*   c2     = (h16*)(ws + alloc((size_t)BT*256*2));
  float* zf     = (float*)(ws + alloc((size_t)BT*64*4));
  float* dconst = (float*)(ws + alloc(256*4));
  float* acc    = (float*)(ws + alloc(256));

  auto gblocks = [](int M, int N) { return ((M >> 4) * (N >> 6) + 7) / 8; };

  init_acc_k<<<1, 1, 0, stream>>>(acc);
  // weight conversion/transposition to fp16 [N][K]
  transpose_h_k<<<256,  256, 0, stream>>>(mlp_w2, mlp2_t, 256, 256);
  transpose_h_k<<<1536, 256, 0, stream>>>(qkv_w,  qkv_t,  256, 1536);
  transpose_h_k<<<64,   256, 0, stream>>>(proj_w, proj_t, 256, 64);
  transpose_h_k<<<256,  256, 0, stream>>>(g1_w,   g1_t,   256, 256);
  transpose_h_k<<<256,  256, 0, stream>>>(g2_w,   g2_t,   256, 256);
  convw_h_k<<<768, 256, 0, stream>>>(ec1_w, ec1_t);
  convw_h_k<<<768, 256, 0, stream>>>(ec2_w, ec2_t);
  convw_h_k<<<768, 256, 0, stream>>>(dc1_w, dc1_t);
  convw_h_k<<<768, 256, 0, stream>>>(dc2_w, dc2_t);
  // encoder
  mlp1_k<<<ROWS, 256, 0, stream>>>(x, mlp_w1, mlp_b1, h1);
  gemm_wmma<true,true,true><<<gblocks(ROWS,256), 256, 0, stream>>>(h1, mlp2_t, mlp_b2, h2, ROWS, 256, 256);
  enc_attn<<<BT, 256, 0, stream>>>(h2, qkv_t, out_w, out_b, hbar);
  im2col_f32_k<<<12288, 256, 0, stream>>>(hbar, ic);
  gemm_wmma<true,true,true><<<gblocks(BT,256), 256, 0, stream>>>(ic, ec1_t, ec1_b, c1, BT, 256, 768);
  im2col_f16_k<<<12288, 256, 0, stream>>>(c1, ic);
  gemm_wmma<true,true,true><<<gblocks(BT,256), 256, 0, stream>>>(ic, ec2_t, ec2_b, c2, BT, 256, 768);
  gemm_wmma<false,true,false><<<gblocks(BT,64), 256, 0, stream>>>(c2, proj_t, proj_b, zf, BT, 64, 256);
  // VQ loss (quantized == codebook row 0 everywhere)
  vqloss_k<<<1024, 256, 0, stream>>>(zf, cb, acc);
  // decoder (z_q constant -> dconst)
  dconst_k<<<1, 256, 0, stream>>>(cb, di_w, di_b, dconst);
  im2col_const_k<<<12288, 256, 0, stream>>>(dconst, ic);
  gemm_wmma<true,true,true><<<gblocks(BT,256), 256, 0, stream>>>(ic, dc1_t, dc1_b, c1, BT, 256, 768);
  im2col_f16_k<<<12288, 256, 0, stream>>>(c1, ic);
  gemm_wmma<true,true,true><<<gblocks(BT,256), 256, 0, stream>>>(ic, dc2_t, dc2_b, c2, BT, 256, 768);
  seedadd_k<<<ROWS, 256, 0, stream>>>(c2, seeds, h1);
  gemm_wmma<true,true,true><<<gblocks(ROWS,256), 256, 0, stream>>>(h1, g1_t, g1_b, h2, ROWS, 256, 256);
  gemm_wmma<true,true,true><<<gblocks(ROWS,256), 256, 0, stream>>>(h2, g2_t, g2_b, h1, ROWS, 256, 256);
  gen3_k<<<RECON_ELEMS/256, 256, 0, stream>>>(h1, g3_w, g3_b, out);
  finalize_k<<<1, 1, 0, stream>>>(acc, out + RECON_ELEMS);
}
